// RegNet_v2_13898514170687
// MI455X (gfx1250) — matrix operand order repaired and checked by reading, not compiled, not run on hardware
//
#include <hip/hip_runtime.h>
#include <hip/hip_bf16.h>
#include <stdint.h>

// ---------------------------------------------------------------------------
// Problem constants (from reference)
// ---------------------------------------------------------------------------
#define B_    4
#define N_    8192
#define NP1_  2048
#define NP2_  512
#define K_    32
#define KQ_   16
#define HW_   4096
#define CRGB_ 64

typedef __attribute__((ext_vector_type(16))) _Float16 v16h;
typedef __attribute__((ext_vector_type(8)))  _Float16 v8h;
typedef __attribute__((ext_vector_type(4)))  _Float16 v4h;
typedef __attribute__((ext_vector_type(8)))  float    v8f;
typedef __attribute__((ext_vector_type(2)))  float    v2f;

#define ASYNC_WAIT(n) asm volatile("s_wait_asynccnt " #n ::: "memory")

// convert 4 contiguous f32 -> packed v4h store (dst must be 8B aligned)
static __device__ __forceinline__ void cvt_store4(_Float16* __restrict__ dst,
                                                  const float* __restrict__ src) {
  v4h h;
  h[0] = (_Float16)src[0]; h[1] = (_Float16)src[1];
  h[2] = (_Float16)src[2]; h[3] = (_Float16)src[3];
  *(v4h*)dst = h;
}

// ---------------------------------------------------------------------------
// WMMA fragment helpers (wave32, 16x16 tiles, contiguous k = 16*(lane/16) + s in both A and B)
// ---------------------------------------------------------------------------

// 16-bit A matrix 16x32 (MxK) from LDS: lane<16 rows hold K 0-15,
// lanes 16-31 hold K 16-31.  Two contiguous 16B runs per lane.
static __device__ __forceinline__ v16h ldsA_frag(const _Float16* __restrict__ Arow0,
                                                 int stride, int k0, int lane) {
  const int m    = lane & 15;
  const int koff = k0 + ((lane & 16) ? 16 : 0);
  const _Float16* p = Arow0 + m * stride + koff;
  v8h lo = *(const v8h*)(p);
  v8h hi = *(const v8h*)(p + 8);
  return __builtin_shufflevector(lo, hi, 0, 1, 2, 3, 4, 5, 6, 7,
                                 8, 9, 10, 11, 12, 13, 14, 15);
}

// 16-bit B matrix 32x16 (KxN) from pre-packed f16 transposed weights
// Wt[cout][kpad]: lanes 0-15 -> K k0..k0+15, lanes 16-31 -> K k0+16..k0+31.
static __device__ __forceinline__ v16h gblB_frag(const _Float16* __restrict__ Wt,
                                                 int kpad, int k0, int n0, int lane) {
  const int n  = n0 + (lane & 15);
  const int kb = k0 + ((lane & 16) ? 16 : 0);
  return *(const v16h*)(Wt + (size_t)n * kpad + kb);
}

// One fused linear+bias+ReLU layer over `rows` LDS rows (rows % 16 == 0),
// executed by a 256-thread (8-wave) block.  IN zero-padded to kpad columns.
static __device__ void mlp_layer(const _Float16* __restrict__ IN, int instride,
                                 _Float16* __restrict__ OUT, int outstride,
                                 const _Float16* __restrict__ Wt, int kpad,
                                 const float* __restrict__ Bias,
                                 int cout, int rows, int tid) {
  const int lane   = tid & 31;
  const int wave   = tid >> 5;
  const int ntiles = rows >> 4;
  for (int tile = wave; tile < ntiles; tile += 8) {
    const int row0 = tile << 4;
    for (int n0 = 0; n0 < cout; n0 += 16) {
      v8f c = {};
      for (int k0 = 0; k0 < kpad; k0 += 32) {
        v16h a  = ldsA_frag(IN + row0 * instride, instride, k0, lane);
        v16h bf = gblB_frag(Wt, kpad, k0, n0, lane);
        c = __builtin_amdgcn_wmma_f32_16x16x32_f16(false, a, false, bf,
                                                   (short)0, c, false, false);
      }
      const int   n    = n0 + (lane & 15);
      const float bias = Bias[n];
      const int   mb   = row0 + ((lane & 16) ? 8 : 0);
#pragma unroll
      for (int r = 0; r < 8; ++r) {
        float v = c[r] + bias;
        v = v > 0.f ? v : 0.f;                       // fused ReLU
        OUT[(mb + r) * outstride + n] = (_Float16)v;
      }
    }
  }
}

// ---------------------------------------------------------------------------
// Weight packing: f32 row-major [cin][cout] -> f16 transposed [cout][kpad],
// zero-padded in K.  permute3!=0 moves the first 3 input channels (rel-xyz)
// to the end, matching the vector-friendly gather layout [features, rel].
// ---------------------------------------------------------------------------
__global__ void pack_w_kernel(const float* __restrict__ W, _Float16* __restrict__ Wt,
                              int cin, int cout, int kpad, int permute3) {
  const int i = blockIdx.x * blockDim.x + threadIdx.x;
  if (i >= cout * kpad) return;
  const int n = i / kpad, k = i % kpad;
  float v = 0.0f;
  if (k < cin) {
    int sk = k;
    if (permute3) {
      const int nf = cin - 3;
      sk = (k < nf) ? (k + 3) : (k - nf);
    }
    v = W[sk * cout + n];
  }
  Wt[i] = (_Float16)v;
}

// ---------------------------------------------------------------------------
// Brute-force kNN via f32 WMMA distance tiles.
// Rank-preserving distance: d' = ||r||^2 - 2 q.r  (||q||^2 dropped: constant
// per query row).  One 64-thread block = 2 waves = 32 queries.
// Refs streamed in 128-point chunks, double-buffered through LDS with
// CDNA5 async loads (global_load_async_to_lds_b32 + s_wait_asynccnt).
// ---------------------------------------------------------------------------
template <int KNN, int DIM>
__global__ __launch_bounds__(64) void knn_kernel(
    const float* __restrict__ qpts, int qbatch, int qstep, int nq,
    const float* __restrict__ rpts, int rbatch, int rstep, int nrefs,
    int* __restrict__ outidx) {
  __shared__ float dist[32 * 128];
  __shared__ float rstage[2][128 * 3];
  const int b     = blockIdx.y;
  const int qbase = blockIdx.x * 32;
  const int tid   = threadIdx.x;
  const int lane  = tid & 31;
  const int wave  = tid >> 5;
  const float* Q = qpts + (size_t)b * qbatch;
  const float* R = rpts + (size_t)b * rbatch;

  // A' fragment (16x4 f32): lane<16 -> K0,K1 ; lane>=16 -> K2,K3
  const int    qm = qbase + wave * 16 + (lane & 15);
  const float* qp = Q + (size_t)qm * qstep * 3;
  v2f afrag;
  if (DIM == 3) {
    if (lane < 16) { afrag[0] = -2.f * qp[0]; afrag[1] = -2.f * qp[1]; }
    else           { afrag[0] = -2.f * qp[2]; afrag[1] = 1.f; }
  } else {
    if (lane < 16) { afrag[0] = -2.f * qp[0]; afrag[1] = -2.f * qp[1]; }
    else           { afrag[0] = 1.f;          afrag[1] = 0.f; }
  }

  // async-stage one 128-point chunk (384 dwords, 6 per thread) into LDS buf
  auto stage = [&](int buf, int r0) {
#pragma unroll
    for (int t = 0; t < 6; ++t) {
      const int e  = t * 64 + tid;
      const int pt = e / 3, c = e % 3;
      const float* g = R + (size_t)(r0 + pt) * rstep * 3 + c;
      const unsigned lds_off = (unsigned)(uintptr_t)&rstage[buf][e];
      asm volatile("global_load_async_to_lds_b32 %0, %1, off"
                   :: "v"(lds_off), "v"((unsigned long long)(uintptr_t)g)
                   : "memory");
    }
  };

  float bd[KNN];
  int   bi[KNN];
#pragma unroll
  for (int i = 0; i < KNN; ++i) { bd[i] = 3.4e38f; bi[i] = 0; }

  stage(0, 0);
  int cur = 0;
  for (int r0 = 0; r0 < nrefs; r0 += 128) {
    if (r0 + 128 < nrefs) {
      stage(cur ^ 1, r0 + 128);  // prefetch next chunk while computing this one
      ASYNC_WAIT(0x6);           // wait current chunk only; next stays in flight
    } else {
      ASYNC_WAIT(0x0);
    }
    __syncthreads();

    // 8 x (16 queries x 16 refs) f32 WMMA tiles -> LDS
    const float* rs = rstage[cur];
#pragma unroll
    for (int t = 0; t < 8; ++t) {
      const float* rp = rs + (t * 16 + (lane & 15)) * 3;
      const float rx = rp[0], ry = rp[1], rz = (DIM == 3) ? rp[2] : 0.f;
      const float rr = rx * rx + ry * ry + rz * rz;
      v2f bfrag;   // B' (4x16 f32): lane<16 -> K0,K1 ; lane>=16 -> K2,K3
      if (DIM == 3) {
        if (lane < 16) { bfrag[0] = rx; bfrag[1] = ry; }
        else           { bfrag[0] = rz; bfrag[1] = rr; }
      } else {
        if (lane < 16) { bfrag[0] = rx; bfrag[1] = ry; }
        else           { bfrag[0] = rr; bfrag[1] = 0.f; }
      }
      v8f c = {};
      c = __builtin_amdgcn_wmma_f32_16x16x4_f32(false, afrag, false, bfrag,
                                                (short)0, c, false, false);
      const int n  = t * 16 + (lane & 15);
      const int mb = wave * 16 + ((lane & 16) ? 8 : 0);
#pragma unroll
      for (int r = 0; r < 8; ++r) dist[(mb + r) * 128 + n] = c[r];
    }
    __syncthreads();

    // one thread per query: register insertion top-K over the 128 candidates
    if (tid < 32) {
      for (int j = 0; j < 128; ++j) {
        const float d = dist[tid * 128 + j];
        if (d < bd[KNN - 1]) {
          int p = KNN - 1;
          while (p > 0 && bd[p - 1] > d) {
            bd[p] = bd[p - 1]; bi[p] = bi[p - 1]; --p;
          }
          bd[p] = d; bi[p] = r0 + j;
        }
      }
    }
    __syncthreads();
    cur ^= 1;
  }

  if (tid < 32) {
    const int q = qbase + tid;
    for (int i = 0; i < KNN; ++i)
      outidx[((size_t)b * nq + q) * KNN + i] = bi[i];
  }
}

// ---------------------------------------------------------------------------
// Fused set-abstraction stage: gather([feat ++ rel-xyz]) -> 3-layer f16-WMMA
// MLP -> max-pool over k.  Block = 4 queries x 32 neighbors = 128 rows.
// Layer-1 weights are packed with the matching channel permutation.
// ---------------------------------------------------------------------------
template <int FEATC, int C1, int C2, int C3, int KP0>
__global__ __launch_bounds__(256) void sa_mlp_kernel(
    const float* __restrict__ xyz, int refstep, int qstep, int nq,
    const float* __restrict__ feat,           // [B][N_/refstep][FEATC]
    const int* __restrict__ idx,              // [B][nq][K_]
    const _Float16* __restrict__ w1, const float* __restrict__ b1,
    const _Float16* __restrict__ w2, const float* __restrict__ b2,
    const _Float16* __restrict__ w3, const float* __restrict__ b3,
    float* __restrict__ outF) {               // [B][nq][C3]
  __shared__ _Float16 Abuf[128 * 64];
  __shared__ _Float16 Bbuf[128 * 64];
  const int b   = blockIdx.y;
  const int q0  = blockIdx.x * 4;
  const int tid = threadIdx.x;
  const float* X = xyz + (size_t)b * (N_ * 3);

  {
    v8h z = {};
    for (int i = tid; i < 128 * 64 / 8; i += 256) {
      ((v8h*)Abuf)[i] = z; ((v8h*)Bbuf)[i] = z;
    }
  }
  __syncthreads();

  // gather stage: [feat(FEATC) ++ rel(3)], vectorized when FEATC % 4 == 0
  for (int r = tid; r < 128; r += 256) {
    const int ql = r >> 5, j = r & 31;
    const int q  = q0 + ql;
    const int id = idx[((size_t)b * nq + q) * K_ + j];
    const float* qp = X + (size_t)q * qstep * 3;
    const float* np = X + (size_t)id * refstep * 3;
    _Float16* Ar = Abuf + r * 64;
    const float* fp = feat + ((size_t)b * (N_ / refstep) + id) * FEATC;
    if (FEATC % 4 == 0) {
#pragma unroll
      for (int c = 0; c < FEATC; c += 4) cvt_store4(Ar + c, fp + c);
    } else {
#pragma unroll
      for (int c = 0; c < FEATC; ++c) Ar[c] = (_Float16)fp[c];
    }
    Ar[FEATC + 0] = (_Float16)(np[0] - qp[0]);
    Ar[FEATC + 1] = (_Float16)(np[1] - qp[1]);
    Ar[FEATC + 2] = (_Float16)(np[2] - qp[2]);
  }
  __syncthreads();

  mlp_layer(Abuf, 64, Bbuf, 64, w1, KP0, b1, C1, 128, tid); __syncthreads();
  mlp_layer(Bbuf, 64, Abuf, 64, w2, 32,  b2, C2, 128, tid); __syncthreads();
  mlp_layer(Abuf, 64, Bbuf, 64, w3, 32,  b3, C3, 128, tid); __syncthreads();

  // max-pool over the 32 neighbors of each query, 4 channels per thread
  for (int w = tid; w < 4 * (C3 / 4); w += 256) {
    const int ql = w / (C3 / 4);
    const int cg = (w % (C3 / 4)) * 4;
    float m0 = -3.4e38f, m1 = m0, m2 = m0, m3 = m0;
    for (int j = 0; j < 32; ++j) {
      v4h v = *(const v4h*)(Bbuf + (ql * 32 + j) * 64 + cg);
      m0 = fmaxf(m0, (float)v[0]); m1 = fmaxf(m1, (float)v[1]);
      m2 = fmaxf(m2, (float)v[2]); m3 = fmaxf(m3, (float)v[3]);
    }
    float* o = outF + ((size_t)b * nq + (q0 + ql)) * C3 + cg;
    o[0] = m0; o[1] = m1; o[2] = m2; o[3] = m3;
  }
}

// ---------------------------------------------------------------------------
// uv projection of p2 (= xyz[::16])
// ---------------------------------------------------------------------------
__global__ void uv_kernel(const float* __restrict__ xyz, float* __restrict__ uvb) {
  const int i = blockIdx.x * blockDim.x + threadIdx.x;
  if (i >= B_ * NP2_) return;
  const int b = i / NP2_, s = i % NP2_;
  const float* p = xyz + ((size_t)b * N_ + (size_t)s * 16) * 3;
  const float z   = p[2];
  const float inv = 1.0f / (fabsf(z) + 1.0f);
  float* u = uvb + (size_t)i * 3;
  u[0] = p[0] * inv; u[1] = p[1] * inv; u[2] = z;
}

__global__ void init_pooled_kernel(float* __restrict__ pooled) {
  const int i = blockIdx.x * blockDim.x + threadIdx.x;
  if (i < B_ * 128) pooled[i] = 0.0f;
}

// ---------------------------------------------------------------------------
// Fused cost-volume MLP + attention + cost + flow MLP + global max-pool.
// Block = 8 queries x 16 px-neighbors = 128 rows, 8 waves.
// Layer-1 channel order: [f2(0..63) ++ rgb(64..127) ++ rel(128..130)],
// matching permuted packed weights -> all feature stores are vector v4h.
// ---------------------------------------------------------------------------
__global__ __launch_bounds__(256) void cv_flow_kernel(
    const float* __restrict__ uvb,   // [B][512][3]
    const float* __restrict__ pix,   // [B][4096][3]
    const float* __restrict__ rgb,   // [B][4096][64]
    const float* __restrict__ f2,    // [B][512][64]
    const int* __restrict__ idx,     // [B][512][16]
    const _Float16* __restrict__ cw0, const float* __restrict__ cb0,
    const _Float16* __restrict__ cw1, const float* __restrict__ cb1,
    const _Float16* __restrict__ cw2, const float* __restrict__ cb2,
    const float* __restrict__ sw,  const float* __restrict__ sb,
    const _Float16* __restrict__ fw0, const float* __restrict__ fb0,
    const _Float16* __restrict__ fw1, const float* __restrict__ fb1,
    float* __restrict__ pooled) {    // [B][128]
  __shared__ _Float16 Abuf[128 * 160];
  __shared__ _Float16 Hbuf[128 * 160];   // fully overwritten before reads
  __shared__ _Float16 Ff[16 * 160];
  __shared__ _Float16 Gg[16 * 160];      // fully overwritten before reads
  __shared__ float sc[128];
  __shared__ float at[128];
  __shared__ float cost[8 * 64];
  const int b   = blockIdx.y;
  const int q0  = blockIdx.x * 8;
  const int tid = threadIdx.x;

  {
    v8h z = {};
    for (int i = tid; i < 128 * 160 / 8; i += 256) ((v8h*)Abuf)[i] = z;
    for (int i = tid; i < 16 * 160 / 8; i += 256) ((v8h*)Ff)[i] = z;
  }
  __syncthreads();

  // gather: 2 threads per row; h=0 -> f2(cols 0..63) + rel(128..130),
  //                            h=1 -> rgb(cols 64..127)
  {
    const int r = tid >> 1;     // 0..127
    const int h = tid & 1;
    const int ql = r >> 4, j = r & 15;
    const int q  = q0 + ql;
    const int id = idx[((size_t)b * NP2_ + q) * KQ_ + j];
    _Float16* Ar = Abuf + r * 160;
    if (h == 0) {
      const float* fq = f2 + ((size_t)b * NP2_ + q) * 64;
#pragma unroll
      for (int c = 0; c < 64; c += 4) cvt_store4(Ar + c, fq + c);
      const float* pu = pix + ((size_t)b * HW_ + id) * 3;
      const float* uq = uvb + ((size_t)b * NP2_ + q) * 3;
      Ar[128] = (_Float16)(pu[0] - uq[0]);
      Ar[129] = (_Float16)(pu[1] - uq[1]);
      Ar[130] = (_Float16)(pu[2] - uq[2]);
    } else {
      const float* rg = rgb + ((size_t)b * HW_ + id) * 64;
#pragma unroll
      for (int c = 0; c < 64; c += 4) cvt_store4(Ar + 64 + c, rg + c);
    }
  }
  __syncthreads();

  mlp_layer(Abuf, 160, Hbuf, 160, cw0, 160, cb0, 128, 128, tid); __syncthreads();
  mlp_layer(Hbuf, 160, Abuf, 160, cw1, 128, cb1,  64, 128, tid); __syncthreads();
  mlp_layer(Abuf, 160, Hbuf, 160, cw2,  64, cb2,  64, 128, tid); __syncthreads();

  // attention scores (64 -> 1), vectorized v4h reads
  if (tid < 128) {
    float s = sb[0];
    const _Float16* hr = Hbuf + tid * 160;
#pragma unroll
    for (int c = 0; c < 64; c += 4) {
      v4h v = *(const v4h*)(hr + c);
      s += (float)v[0] * sw[c]     + (float)v[1] * sw[c + 1] +
           (float)v[2] * sw[c + 2] + (float)v[3] * sw[c + 3];
    }
    sc[tid] = s;
  }
  __syncthreads();

  // softmax over the 16 neighbors of each query
  if (tid < 8) {
    float mx = -3.4e38f;
    for (int j = 0; j < 16; ++j) mx = fmaxf(mx, sc[tid * 16 + j]);
    float e[16], sum = 0.f;
    for (int j = 0; j < 16; ++j) { e[j] = __expf(sc[tid * 16 + j] - mx); sum += e[j]; }
    for (int j = 0; j < 16; ++j) at[tid * 16 + j] = e[j] / sum;
  }
  __syncthreads();

  // cost = sum_k h * attn, 4 channels per thread
  for (int w = tid; w < 8 * 16; w += 256) {
    const int ql = w >> 4, cg = (w & 15) * 4;
    float s0 = 0.f, s1 = 0.f, s2 = 0.f, s3 = 0.f;
    for (int j = 0; j < 16; ++j) {
      const float a = at[ql * 16 + j];
      v4h v = *(const v4h*)(Hbuf + (ql * 16 + j) * 160 + cg);
      s0 += (float)v[0] * a; s1 += (float)v[1] * a;
      s2 += (float)v[2] * a; s3 += (float)v[3] * a;
    }
    float* cp = cost + ql * 64 + cg;
    cp[0] = s0; cp[1] = s1; cp[2] = s2; cp[3] = s3;
  }
  __syncthreads();

  // flow input rows: [f2(64) ++ cost(64)] (8 valid rows of a 16-row tile)
  for (int w = tid; w < 8 * 32; w += 256) {
    const int ql = w >> 5, cg = (w & 31) * 4;
    const float* src = (cg < 64)
        ? f2 + ((size_t)b * NP2_ + (q0 + ql)) * 64 + cg
        : cost + ql * 64 + (cg - 64);
    cvt_store4(Ff + ql * 160 + cg, src);
  }
  __syncthreads();

  mlp_layer(Ff, 160, Gg, 160, fw0, 128, fb0, 128, 16, tid); __syncthreads();
  mlp_layer(Gg, 160, Ff, 160, fw1, 128, fb1,  64, 16, tid); __syncthreads();

  // global max-pool of [f2 ++ flow] (all >= 0 after ReLU -> uint-bit atomicMax)
  for (int w = tid; w < 8 * 32; w += 256) {
    const int ql = w >> 5, cg = (w & 31) * 4;
    float v0, v1, v2, v3;
    if (cg < 64) {
      const float* fq = f2 + ((size_t)b * NP2_ + (q0 + ql)) * 64 + cg;
      v0 = fq[0]; v1 = fq[1]; v2 = fq[2]; v3 = fq[3];
    } else {
      v4h v = *(const v4h*)(Ff + ql * 160 + (cg - 64));
      v0 = (float)v[0]; v1 = (float)v[1]; v2 = (float)v[2]; v3 = (float)v[3];
    }
    unsigned int* pp = (unsigned int*)&pooled[b * 128 + cg];
    atomicMax(pp + 0, __float_as_uint(v0));
    atomicMax(pp + 1, __float_as_uint(v1));
    atomicMax(pp + 2, __float_as_uint(v2));
    atomicMax(pp + 3, __float_as_uint(v3));
  }
}

// ---------------------------------------------------------------------------
// Tiny head: pooled(128) -> relu 256 -> {q(4) normalized, t(3)}
// ---------------------------------------------------------------------------
__global__ __launch_bounds__(256) void head_kernel(
    const float* __restrict__ pooled,
    const float* __restrict__ hw, const float* __restrict__ hb,
    const float* __restrict__ qw, const float* __restrict__ qb,
    const float* __restrict__ tw, const float* __restrict__ tb,
    float* __restrict__ out) {
  __shared__ float pb[128];
  __shared__ float hid[256];
  __shared__ float o[7];
  const int tid = threadIdx.x;
  for (int b = 0; b < B_; ++b) {
    if (tid < 128) pb[tid] = pooled[b * 128 + tid];
    __syncthreads();
    float s = hb[tid];
    for (int c = 0; c < 128; ++c) s += pb[c] * hw[c * 256 + tid];
    hid[tid] = fmaxf(s, 0.f);
    __syncthreads();
    if (tid < 7) {
      float acc;
      if (tid < 4) {
        acc = qb[tid];
        for (int k = 0; k < 256; ++k) acc += hid[k] * qw[k * 4 + tid];
      } else {
        acc = tb[tid - 4];
        for (int k = 0; k < 256; ++k) acc += hid[k] * tw[k * 3 + (tid - 4)];
      }
      o[tid] = acc;
    }
    __syncthreads();
    if (tid == 0) {
      const float nrm =
          sqrtf(o[0]*o[0] + o[1]*o[1] + o[2]*o[2] + o[3]*o[3]) + 1e-8f;
      out[b * 7 + 0] = o[0] / nrm; out[b * 7 + 1] = o[1] / nrm;
      out[b * 7 + 2] = o[2] / nrm; out[b * 7 + 3] = o[3] / nrm;
      out[b * 7 + 4] = o[4];       out[b * 7 + 5] = o[5];
      out[b * 7 + 6] = o[6];
    }
    __syncthreads();
  }
}

// ---------------------------------------------------------------------------
// Launch
// ---------------------------------------------------------------------------
extern "C" void kernel_launch(void* const* d_in, const int* in_sizes, int n_in,
                              void* d_out, int out_size, void* d_ws, size_t ws_size,
                              hipStream_t stream) {
  (void)in_sizes; (void)n_in; (void)out_size; (void)ws_size;

  const float* lidar_xyz  = (const float*)d_in[0];
  const float* lidar_feat = (const float*)d_in[1];
  const float* rgb_feat   = (const float*)d_in[2];
  const float* pixel_uv   = (const float*)d_in[3];

  // params flattened in pytree (sorted-dict-key) order:
  // cv, cv_score, flow, head, head_q, head_t, sa1, sa2
  int p = 4;
  const float* cv_w0 = (const float*)d_in[p++]; const float* cv_b0 = (const float*)d_in[p++];
  const float* cv_w1 = (const float*)d_in[p++]; const float* cv_b1 = (const float*)d_in[p++];
  const float* cv_w2 = (const float*)d_in[p++]; const float* cv_b2 = (const float*)d_in[p++];
  const float* cvs_w = (const float*)d_in[p++]; const float* cvs_b = (const float*)d_in[p++];
  const float* fl_w0 = (const float*)d_in[p++]; const float* fl_b0 = (const float*)d_in[p++];
  const float* fl_w1 = (const float*)d_in[p++]; const float* fl_b1 = (const float*)d_in[p++];
  const float* hd_w  = (const float*)d_in[p++]; const float* hd_b  = (const float*)d_in[p++];
  const float* hq_w  = (const float*)d_in[p++]; const float* hq_b  = (const float*)d_in[p++];
  const float* ht_w  = (const float*)d_in[p++]; const float* ht_b  = (const float*)d_in[p++];
  const float* s1_w0 = (const float*)d_in[p++]; const float* s1_b0 = (const float*)d_in[p++];
  const float* s1_w1 = (const float*)d_in[p++]; const float* s1_b1 = (const float*)d_in[p++];
  const float* s1_w2 = (const float*)d_in[p++]; const float* s1_b2 = (const float*)d_in[p++];
  const float* s2_w0 = (const float*)d_in[p++]; const float* s2_b0 = (const float*)d_in[p++];
  const float* s2_w1 = (const float*)d_in[p++]; const float* s2_b1 = (const float*)d_in[p++];
  const float* s2_w2 = (const float*)d_in[p++]; const float* s2_b2 = (const float*)d_in[p++];

  // workspace carve-out
  uint8_t* ws  = (uint8_t*)d_ws;
  size_t   off = 0;
  auto salloc = [&](size_t bytes) -> void* {
    void* r = ws + off;
    off = (off + bytes + 255) & ~(size_t)255;
    return r;
  };
  int*   idx1   = (int*)  salloc((size_t)B_ * NP1_ * K_  * 4);
  float* f1     = (float*)salloc((size_t)B_ * NP1_ * 32  * 4);
  int*   idx2   = (int*)  salloc((size_t)B_ * NP2_ * K_  * 4);
  float* f2     = (float*)salloc((size_t)B_ * NP2_ * 64  * 4);
  float* uvb    = (float*)salloc((size_t)B_ * NP2_ * 3   * 4);
  int*   idx3   = (int*)  salloc((size_t)B_ * NP2_ * KQ_ * 4);
  float* pooled = (float*)salloc((size_t)B_ * 128 * 4);

  // packed f16-transposed weights [cout][kpad]
  auto wt_alloc = [&](int cout, int kpad) -> _Float16* {
    return (_Float16*)salloc((size_t)cout * kpad * 2);
  };
  _Float16* t_s1w0 = wt_alloc(16, 32);   _Float16* t_s1w1 = wt_alloc(16, 32);
  _Float16* t_s1w2 = wt_alloc(32, 32);
  _Float16* t_s2w0 = wt_alloc(32, 64);   _Float16* t_s2w1 = wt_alloc(32, 32);
  _Float16* t_s2w2 = wt_alloc(64, 32);
  _Float16* t_cvw0 = wt_alloc(128, 160); _Float16* t_cvw1 = wt_alloc(64, 128);
  _Float16* t_cvw2 = wt_alloc(64, 64);
  _Float16* t_flw0 = wt_alloc(128, 128); _Float16* t_flw1 = wt_alloc(64, 128);

  auto pack = [&](const float* W, _Float16* Wt, int cin, int cout, int kpad, int perm) {
    const int n = cout * kpad;
    pack_w_kernel<<<dim3((n + 255) / 256), dim3(256), 0, stream>>>(W, Wt, cin, cout, kpad, perm);
  };
  // layer-1 matrices use the [features, rel] permutation (rel moved to end)
  pack(s1_w0, t_s1w0, 6, 16, 32, 1);   pack(s1_w1, t_s1w1, 16, 16, 32, 0);
  pack(s1_w2, t_s1w2, 16, 32, 32, 0);
  pack(s2_w0, t_s2w0, 35, 32, 64, 1);  pack(s2_w1, t_s2w1, 32, 32, 32, 0);
  pack(s2_w2, t_s2w2, 32, 64, 32, 0);
  pack(cv_w0, t_cvw0, 131, 128, 160, 1); pack(cv_w1, t_cvw1, 128, 64, 128, 0);
  pack(cv_w2, t_cvw2, 64, 64, 64, 0);
  pack(fl_w0, t_flw0, 128, 128, 128, 0); pack(fl_w1, t_flw1, 128, 64, 128, 0);

  init_pooled_kernel<<<dim3(2), dim3(256), 0, stream>>>(pooled);

  // SA1: 2048 queries (xyz stride 4) vs 8192 refs
  knn_kernel<32, 3><<<dim3(NP1_ / 32, B_), dim3(64), 0, stream>>>(
      lidar_xyz, N_ * 3, 4, NP1_, lidar_xyz, N_ * 3, 1, N_, idx1);
  sa_mlp_kernel<3, 16, 16, 32, 32><<<dim3(NP1_ / 4, B_), dim3(256), 0, stream>>>(
      lidar_xyz, 1, 4, NP1_, lidar_feat, idx1,
      t_s1w0, s1_b0, t_s1w1, s1_b1, t_s1w2, s1_b2, f1);

  // SA2: 512 queries (xyz stride 16) vs 2048 refs (xyz stride 4)
  knn_kernel<32, 3><<<dim3(NP2_ / 32, B_), dim3(64), 0, stream>>>(
      lidar_xyz, N_ * 3, 16, NP2_, lidar_xyz, N_ * 3, 4, NP1_, idx2);
  sa_mlp_kernel<32, 32, 32, 64, 64><<<dim3(NP2_ / 4, B_), dim3(256), 0, stream>>>(
      lidar_xyz, 4, 16, NP2_, f1, idx2,
      t_s2w0, s2_b0, t_s2w1, s2_b1, t_s2w2, s2_b2, f2);

  // uv projection + pixel kNN (2-D)
  uv_kernel<<<dim3((B_ * NP2_ + 255) / 256), dim3(256), 0, stream>>>(lidar_xyz, uvb);
  knn_kernel<16, 2><<<dim3(NP2_ / 32, B_), dim3(64), 0, stream>>>(
      uvb, NP2_ * 3, 1, NP2_, pixel_uv, HW_ * 3, 1, HW_, idx3);

  // fused cost-volume + attention + flow + global max-pool
  cv_flow_kernel<<<dim3(NP2_ / 8, B_), dim3(256), 0, stream>>>(
      uvb, pixel_uv, rgb_feat, f2, idx3,
      t_cvw0, cv_b0, t_cvw1, cv_b1, t_cvw2, cv_b2,
      cvs_w, cvs_b, t_flw0, fl_b0, t_flw1, fl_b1, pooled);

  head_kernel<<<dim3(1), dim3(256), 0, stream>>>(
      pooled, hd_w, hd_b, hq_w, hq_b, ht_w, ht_b, (float*)d_out);
}
